// ModConv_65154653880362
// MI455X (gfx1250) — matrix operand order repaired and checked by reading, not compiled, not run on hardware
//
#include <hip/hip_runtime.h>
#include <hip/hip_bf16.h>

#define B_   16
#define CIN  128
#define COUT 128
#define SD   512
#define H_   128
#define W_   128

typedef __attribute__((ext_vector_type(16))) __bf16 v16bf;
typedef __attribute__((ext_vector_type(8)))  float  v8f;

static __device__ __forceinline__ unsigned short f2bf(float f) {
    __bf16 h = (__bf16)f;
    return __builtin_bit_cast(unsigned short, h);
}

// ---------------------------------------------------------------------------
// Stage 1: style[b][i] = sum_s sv[b][s]*(1/sqrt(512)) * proj_w[i][s] + proj_b[i]
// ---------------------------------------------------------------------------
__global__ void k_style(const float* __restrict__ sv, const float* __restrict__ pw,
                        const float* __restrict__ pb, float* __restrict__ style) {
    int t = blockIdx.x * blockDim.x + threadIdx.x;   // 0 .. 2047
    int b = t >> 7, i = t & 127;
    const float scale = 0.04419417382415922f;        // 1/sqrt(512)
    float sum = pb[i];
    const float* svb = sv + b * SD;
    const float* pwi = pw + i * SD;
    for (int s = 0; s < SD; ++s) sum += scale * svb[s] * pwi[s];
    style[t] = sum;
}

// ---------------------------------------------------------------------------
// Stage 2: demod[b][o] = rsqrt( sum_{c,kp} (weight[o][c][kp]*style[b][c])^2 + eps )
// One wave per (b,o) pair.
// ---------------------------------------------------------------------------
__global__ void k_demod(const float* __restrict__ weight, const float* __restrict__ style,
                        float* __restrict__ demod) {
    int gtid = blockIdx.x * blockDim.x + threadIdx.x;
    int wv   = gtid >> 5;                            // 0 .. 2047
    int lane = gtid & 31;
    int b = wv >> 7, o = wv & 127;
    float sum = 0.f;
    for (int idx = lane; idx < CIN * 9; idx += 32) {
        int c  = idx / 9;
        int kp = idx - c * 9;
        float v = weight[(o * CIN + c) * 9 + kp] * style[b * CIN + c];
        sum += v * v;
    }
    for (int off = 16; off; off >>= 1) sum += __shfl_xor(sum, off, 32);
    if (lane == 0) demod[b * COUT + o] = rsqrtf(sum + 1e-8f);
}

// ---------------------------------------------------------------------------
// Stage 3: pack modulated weights into WMMA-B fragment-ready bf16 layout.
// Block index: [b][kpos][cchunk][ntile][lane][j]; lane reads its 16 bf16
// (32 contiguous bytes) directly in the conv kernel.
// B layout (32x16 bf16): lane L -> n = L%16, half = L/16, element j -> K = 8*half + (j&7) + 16*(j>>3).
// ---------------------------------------------------------------------------
__global__ void k_wfrag(const float* __restrict__ weight, const float* __restrict__ style,
                        const float* __restrict__ demod, unsigned short* __restrict__ wfrag) {
    int t = blockIdx.x * blockDim.x + threadIdx.x;   // 0 .. 2359295
    int j      = t & 15;
    int lane   = (t >> 4) & 31;
    int ntile  = (t >> 9) & 7;
    int cchunk = (t >> 12) & 3;
    int kpos   = (t >> 14) % 9;
    int b      = (t >> 14) / 9;
    int n    = ntile * 16 + (lane & 15);
    int half = lane >> 4;
    int c    = cchunk * 32 + half * 8 + (j & 7) + (j >> 3) * 16;
    float v = weight[(n * CIN + c) * 9 + kpos] * style[b * CIN + c] * demod[b * COUT + n];
    wfrag[t] = f2bf(v);
}

// ---------------------------------------------------------------------------
// Stage 4: implicit-GEMM conv. One block = one sample, one 64-pixel half-row,
// all 128 output channels. 8 waves = 4 (M-subtiles) x 2 (N-halves); each wave
// owns 4 v8f accumulators (16px x 64ch). K loop: 9 taps x 4 channel-chunks,
// each step one v_wmma_f32_16x16x32_bf16 per accumulator.
// ---------------------------------------------------------------------------
#define XS_ELEMS (3 * 66 * CIN)          // 25344 bf16 = 50688 B
#define SMEM_BYTES (XS_ELEMS * 2)        // > COUT*65*4 = 33280 B

__global__ __launch_bounds__(256) void k_conv(const float* __restrict__ x,
                                              const unsigned short* __restrict__ wfrag,
                                              float* __restrict__ out) {
    __shared__ __align__(32) unsigned char smem[SMEM_BYTES];
    unsigned short* xs   = (unsigned short*)smem;
    float*          outs = (float*)smem;

    int tid  = threadIdx.x;
    int blk  = blockIdx.x;
    int b    = blk >> 8;
    int tile = blk & 255;
    int y    = tile >> 1;
    int x0   = (tile & 1) * 64;

    // --- stage x tile into LDS as bf16, channels permuted to match the
    //     interleaved 16-bit A-matrix K pattern (swap 8-groups 1<->2 per 32) ---
    for (int idx = tid; idx < XS_ELEMS; idx += 256) {
        int c   = idx / 198;
        int rem = idx - c * 198;
        int row = rem / 66;
        int col = rem - row * 66;
        int yy  = y + row - 1;
        int xx  = x0 + col - 1;
        float v = 0.f;
        if ((unsigned)yy < (unsigned)H_ && (unsigned)xx < (unsigned)W_)
            v = x[((b * CIN + c) * H_ + yy) * W_ + xx];
        unsigned cc = (unsigned)c & 31u;
        unsigned g  = (cc >> 3) & 3u;
        unsigned pg = ((g & 1u) << 1) | (g >> 1);           // swap group bits
        unsigned p  = ((unsigned)c & ~31u) | (cc & 7u) | (pg << 3);
        xs[(row * 66 + col) * CIN + p] = f2bf(v);
    }
    __syncthreads();

    int wv    = tid >> 5;
    int lane  = tid & 31;
    int m_sub = wv & 3;       // which 16-pixel subtile
    int nhalf = wv >> 2;      // which 64-channel half
    int half  = lane >> 4;
    int m     = lane & 15;

    v8f acc[4] = {v8f{}, v8f{}, v8f{}, v8f{}};

    for (int kpos = 0; kpos < 9; ++kpos) {
        int ky = kpos / 3;
        int kx = kpos - ky * 3;
        int acol = m_sub * 16 + m + kx;                     // 0..65
        const unsigned short* xrow = xs + (ky * 66 + acol) * CIN + half * 16;
        const unsigned short* wk =
            wfrag + (size_t)(((b * 9 + kpos) * 4) * 8) * 512 + lane * 16;
#pragma unroll
        for (int cc = 0; cc < 4; ++cc) {
            v16bf a = *(const v16bf*)(xrow + cc * 32);
            const unsigned short* wc = wk + (size_t)cc * 8 * 512;
#pragma unroll
            for (int t = 0; t < 4; ++t) {
                v16bf bf = *(const v16bf*)(wc + (size_t)(nhalf * 4 + t) * 512);
                acc[t] = __builtin_amdgcn_wmma_f32_16x16x32_bf16(
                    false, a, false, bf, (short)0, acc[t], false, false);
            }
        }
    }

    __syncthreads();   // done reading xs; reuse smem as output staging

    // D layout: lane L -> n = L%16 (per 16-tile), vgpr v -> m = v + 8*(L/16)
#pragma unroll
    for (int t = 0; t < 4; ++t) {
        int n = (nhalf * 4 + t) * 16 + m;
#pragma unroll
        for (int v = 0; v < 8; ++v) {
            int ml = m_sub * 16 + v + 8 * half;
            outs[n * 65 + ml] = acc[t][v];                  // stride 65: no bank conflict
        }
    }
    __syncthreads();

    for (int idx = tid; idx < COUT * 64; idx += 256) {
        int n  = idx >> 6;
        int ml = idx & 63;
        out[((b * COUT + n) * H_ + y) * W_ + x0 + ml] = outs[n * 65 + ml];
    }
}

// ---------------------------------------------------------------------------
extern "C" void kernel_launch(void* const* d_in, const int* in_sizes, int n_in,
                              void* d_out, int out_size, void* d_ws, size_t ws_size,
                              hipStream_t stream) {
    const float* x  = (const float*)d_in[0];   // [16,128,128,128]
    const float* sv = (const float*)d_in[1];   // [16,512]
    const float* pw = (const float*)d_in[2];   // [128,512]
    const float* pb = (const float*)d_in[3];   // [128]
    const float* w  = (const float*)d_in[4];   // [1,128,128,3,3]
    float* out = (float*)d_out;                // [16,128,128,128]

    float* style = (float*)d_ws;                                   // 2048 f32
    float* demod = style + B_ * CIN;                               // 2048 f32
    unsigned short* wfrag = (unsigned short*)(demod + B_ * COUT);  // 2,359,296 bf16

    k_style<<<8, 256, 0, stream>>>(sv, pw, pb, style);
    k_demod<<<256, 256, 0, stream>>>(w, style, demod);
    k_wfrag<<<9216, 256, 0, stream>>>(w, style, demod, wfrag);
    k_conv<<<B_ * 256, 256, 0, stream>>>(x, wfrag, out);
}
